// GPT_35837207118567
// MI455X (gfx1250) — matrix operand order repaired and checked by reading, not compiled, not run on hardware
//
#include <hip/hip_runtime.h>
#include <hip/hip_bf16.h>

typedef unsigned short u16;
typedef unsigned int   u32;

typedef __attribute__((ext_vector_type(16))) __bf16 v16bf;
typedef __attribute__((ext_vector_type(8)))  float  v8f;
typedef int v4i_ __attribute__((vector_size(16)));

union ABu { u32 u[8]; v16bf b; };

#define B_   2
#define T_   2048
#define C_   768
#define H_   12
#define D_   64
#define C3_  2304
#define BT_  (B_*T_)
#define BTC_ (B_*T_*C_)
#define W3_  (C3_*C_)
#define WP_  (C_*C_)

#if defined(__HIP_DEVICE_COMPILE__) && __has_builtin(__builtin_amdgcn_global_load_async_to_lds_b128)
#define HAVE_ASYNC 1
#else
#define HAVE_ASYNC 0
#endif

__device__ __forceinline__ u16 f2bf(float f) {
  u32 u = __builtin_bit_cast(u32, f);
  u32 r = u + 0x7FFFu + ((u >> 16) & 1u);
  return (u16)(r >> 16);
}
__device__ __forceinline__ float bf2f(u16 h) {
  u32 u = ((u32)h) << 16;
  return __builtin_bit_cast(float, u);
}
__device__ __forceinline__ v8f wmma_bf(const ABu& a, const ABu& b, v8f c) {
  return __builtin_amdgcn_wmma_f32_16x16x32_bf16(false, a.b, false, b.b,
                                                 (short)0, c, false, false);
}
// A operand (16-bit, 16x32), contiguous k order used here: lane holds row=lane%16; VGPR r holds K pair at:
__device__ __forceinline__ int a_koff(int r, int hl) {
  return 2*r + hl*16;
}

// 16-byte global -> LDS copy; async (ASYNCcnt) when available.
__device__ __forceinline__ void async_cp16(const void* g, void* l) {
#if HAVE_ASYNC
  __builtin_amdgcn_global_load_async_to_lds_b128((v4i_*)g, (v4i_*)l, 0, 0);
#else
  union U16B { uint4 v; };
  *(U16B*)l = *(const U16B*)g;
#endif
}

template<int N>
__device__ __forceinline__ void wait_async() {
#if HAVE_ASYNC
#if __has_builtin(__builtin_amdgcn_s_wait_asynccnt)
  __builtin_amdgcn_s_wait_asynccnt(N);
#else
  asm volatile("s_wait_asynccnt %0" :: "i"(N) : "memory");
#endif
#endif
}

// ---------------------------------------------------------------------------
// Preprocess: bf16 casts + center/radius + |W|
// ---------------------------------------------------------------------------
__global__ void prep_x(const float* __restrict__ x, const float* __restrict__ xl,
                       const float* __restrict__ xu,
                       u16* xbf, u16* xcbf, u16* xrbf) {
  int i = blockIdx.x * 256 + threadIdx.x;
  if (i >= BTC_) return;
  xbf[i] = f2bf(x[i]);
  float lo = xl[i], hi = xu[i];
  xcbf[i] = f2bf(0.5f * (lo + hi));
  xrbf[i] = f2bf(0.5f * (hi - lo));
}

__global__ void prep_w(const float* __restrict__ W, const float* __restrict__ Wpj,
                       u16* wbf, u16* wabf, u16* wpbf) {
  int i = blockIdx.x * 256 + threadIdx.x;
  if (i < W3_) { float w = W[i]; wbf[i] = f2bf(w); wabf[i] = f2bf(fabsf(w)); }
  if (i < WP_) { wpbf[i] = f2bf(Wpj[i]); }
}

// ---------------------------------------------------------------------------
// QKV GEMM: nom = x@W^T, cen = xc@W^T, rad = xr@|W|^T ; lo=cen-rad hi=cen+rad
// Software-pipelined: chunk k+1 loads issued before chunk k WMMAs.
// block=128 (4 waves, 16 cols each); grid=(3C/64=36, BT/16=256)
// ---------------------------------------------------------------------------
struct QkvArgs {
  const u16 *xb, *xcb, *xrb, *wb, *wab;
  u16 *nom[3], *lo[3], *hi[3];
};

__device__ __forceinline__ void qkv_load(const QkvArgs& g, size_t arow, size_t brow,
                                         int kc, int hl,
                                         ABu& ax, ABu& axc, ABu& axr,
                                         ABu& bw, ABu& bwa) {
#pragma unroll
  for (int r = 0; r < 8; ++r) {
    int kk = kc * 32 + a_koff(r, hl);
    ax.u[r]  = *(const u32*)(g.xb  + arow + kk);
    axc.u[r] = *(const u32*)(g.xcb + arow + kk);
    axr.u[r] = *(const u32*)(g.xrb + arow + kk);
    int kb = kc * 32 + hl * 16 + 2 * r;
    bw.u[r]  = *(const u32*)(g.wb  + brow + kb);
    bwa.u[r] = *(const u32*)(g.wab + brow + kb);
  }
}

__global__ __launch_bounds__(128) void qkv_gemm(QkvArgs g) {
  int lane = threadIdx.x & 31, wave = threadIdx.x >> 5;
  int hl = lane >> 4, mr = lane & 15;
  int jbase = blockIdx.x * 64 + wave * 16;
  int tbase = blockIdx.y * 16;

  v8f accn = {0,0,0,0,0,0,0,0};
  v8f accc = accn, accr = accn;

  const size_t arow = (size_t)(tbase + mr) * C_;
  const size_t brow = (size_t)(jbase + mr) * C_;

  ABu aA0, aC0, aR0, bW0, bA0;   // ping
  ABu aA1, aC1, aR1, bW1, bA1;   // pong
  qkv_load(g, arow, brow, 0, hl, aA0, aC0, aR0, bW0, bA0);

  const int NK = C_ / 32;        // 24 (even)
  for (int kc = 0; kc < NK; kc += 2) {
    qkv_load(g, arow, brow, kc + 1, hl, aA1, aC1, aR1, bW1, bA1);
    accn = wmma_bf(aA0, bW0, accn);
    accc = wmma_bf(aC0, bW0, accc);
    accr = wmma_bf(aR0, bA0, accr);
    if (kc + 2 < NK)
      qkv_load(g, arow, brow, kc + 2, hl, aA0, aC0, aR0, bW0, bA0);
    accn = wmma_bf(aA1, bW1, accn);
    accc = wmma_bf(aC1, bW1, accc);
    accr = wmma_bf(aR1, bA1, accr);
  }

  int j = jbase + mr;            // column in [0, 3C)
  int s = j / C_;                // 0=q 1=k 2=v
  int jj = j - s * C_;
  int h = jj >> 6, d = jj & 63;
#pragma unroll
  for (int r = 0; r < 8; ++r) {
    int row = tbase + r + 8 * hl;        // flattened b*T + t
    int b = row >> 11, t = row & (T_ - 1);
    size_t o = (((size_t)b * H_ + h) * T_ + t) * D_ + d;
    g.nom[s][o] = f2bf(accn[r]);
    g.lo[s][o]  = f2bf(accc[r] - accr[r]);
    g.hi[s][o]  = f2bf(accc[r] + accr[r]);
  }
}

// ---------------------------------------------------------------------------
// Flash attention. Block = 4 waves x 16 query rows = 64 rows, one (bh,variant).
// Key tiles of 32 staged K/V0/V1 -> LDS via async copies, double buffered.
// Uniform trip count per block (masked tiles contribute exp(-inf)=0).
// grid=(T/64=32, B*H=24, 5 variants); block=128
// ---------------------------------------------------------------------------
struct AttnArgs {
  const u16 *qn, *kn, *vn, *ql, *qh, *kl, *kh, *vl, *vh;
  u16 *ynom; u16 *yc[8];
};

__device__ __forceinline__ void stage_tile(const u16* K, const u16* V0, const u16* V1,
                                           size_t bhT, int kbase,
                                           u16* dK, u16* dV0, u16* dV1, int tid) {
  // each tile is a contiguous 4 KB block (32 rows x 64 bf16)
  const char* sK = (const char*)(K  + (bhT + kbase) * D_);
  const char* s0 = (const char*)(V0 + (bhT + kbase) * D_);
  const char* s1 = (const char*)(V1 + (bhT + kbase) * D_);
#pragma unroll
  for (int i = 0; i < 6; ++i) {
    int j = i * 128 + tid;            // 0..767 (3 tensors x 256 x 16B)
    int e = j >> 8;
    int boff = (j & 255) * 16;
    const char* src = (e == 0) ? sK : ((e == 1) ? s0 : s1);
    char* dst = (char*)((e == 0) ? dK : ((e == 1) ? dV0 : dV1)) + boff;
    async_cp16(src + boff, dst);
  }
}

__global__ __launch_bounds__(128) void attn_fa(AttnArgs g) {
  __shared__ u16 sK[2][32 * 64];
  __shared__ u16 sV0[2][32 * 64];
  __shared__ u16 sV1[2][32 * 64];
  __shared__ u16 sP[4][512];

  int tid = threadIdx.x;
  int lane = tid & 31, wave = tid >> 5;
  int hl = lane >> 4, mr = lane & 15;
  int qbase = (blockIdx.x * 4 + wave) * 16;
  int bh = blockIdx.y;
  int var = blockIdx.z;

  const u16 *Q, *K, *V0, *V1; u16 *O0, *O1 = nullptr;
  if (var == 0) { Q = g.qn; K = g.kn; V0 = g.vn; V1 = g.vn; O0 = g.ynom; }
  else {
    int qi = (var - 1) >> 1, ki = (var - 1) & 1;
    Q = qi ? g.qh : g.ql;  K = ki ? g.kh : g.kl;
    V0 = g.vl; V1 = g.vh;
    O0 = g.yc[qi * 4 + ki * 2];  O1 = g.yc[qi * 4 + ki * 2 + 1];
  }
  const size_t bhT = (size_t)bh * T_;

  // Q tile in A-layout (two K=32 chunks over D=64)
  ABu qa0, qa1;
  {
    const u16* Qp = Q + (bhT + qbase + mr) * D_;
#pragma unroll
    for (int r = 0; r < 8; ++r) {
      int kk = a_koff(r, hl);
      qa0.u[r] = *(const u32*)(Qp + kk);
      qa1.u[r] = *(const u32*)(Qp + 32 + kk);
    }
  }

  const v8f zf = {0,0,0,0,0,0,0,0};
  float m_r[8], l_r[8];
#pragma unroll
  for (int r = 0; r < 8; ++r) { m_r[r] = -1e30f; l_r[r] = 0.f; }
  v8f o0[4], o1[4];
#pragma unroll
  for (int i = 0; i < 4; ++i) { o0[i] = zf; o1[i] = zf; }

  u16* lds = sP[wave];
  const float scale = 0.125f;                // 1/sqrt(64)
  int nkt = blockIdx.x * 2 + 2;              // uniform across the block

  stage_tile(K, V0, V1, bhT, 0, sK[0], sV0[0], sV1[0], tid);

  for (int kt = 0; kt < nkt; ++kt) {
    int cur = kt & 1;
    if (kt + 1 < nkt) {
      stage_tile(K, V0, V1, bhT, (kt + 1) * 32, sK[cur ^ 1], sV0[cur ^ 1], sV1[cur ^ 1], tid);
      wait_async<6>();                       // current tile's 6 copies done; prefetch in flight
    } else {
      wait_async<0>();
    }
    __syncthreads();

    int kbase = kt * 32;
    const u16* ldsK  = sK[cur];
    const u16* ldsV0 = sV0[cur];
    const u16* ldsV1 = sV1[cur];

    v8f s0 = zf, s1 = zf;
    {
      ABu kb0, kb1;
#pragma unroll
      for (int r = 0; r < 8; ++r) {
        int dOf = hl * 16 + 2 * r;
        kb0.u[r] = *(const u32*)(ldsK + mr * 64 + dOf);
        kb1.u[r] = *(const u32*)(ldsK + mr * 64 + 32 + dOf);
      }
      s0 = wmma_bf(qa0, kb0, s0);
      s0 = wmma_bf(qa1, kb1, s0);
#pragma unroll
      for (int r = 0; r < 8; ++r) {
        int dOf = hl * 16 + 2 * r;
        kb0.u[r] = *(const u32*)(ldsK + (16 + mr) * 64 + dOf);
        kb1.u[r] = *(const u32*)(ldsK + (16 + mr) * 64 + 32 + dOf);
      }
      s1 = wmma_bf(qa0, kb0, s1);
      s1 = wmma_bf(qa1, kb1, s1);
    }

    float alpha[8];
#pragma unroll
    for (int r = 0; r < 8; ++r) {
      int q  = qbase + r + 8 * hl;
      int k0 = kbase + mr, k1 = kbase + 16 + mr;
      s0[r] = (k0 <= q) ? s0[r] * scale : -1e30f;
      s1[r] = (k1 <= q) ? s1[r] * scale : -1e30f;
      float mx = fmaxf(s0[r], s1[r]);
      mx = fmaxf(mx, __shfl_xor(mx, 1));
      mx = fmaxf(mx, __shfl_xor(mx, 2));
      mx = fmaxf(mx, __shfl_xor(mx, 4));
      mx = fmaxf(mx, __shfl_xor(mx, 8));
      float mn = fmaxf(m_r[r], mx);
      alpha[r] = __expf(m_r[r] - mn);
      m_r[r] = mn;
      float p0 = __expf(s0[r] - mn), p1 = __expf(s1[r] - mn);
      s0[r] = p0; s1[r] = p1;
      float rs = p0 + p1;
      rs += __shfl_xor(rs, 1);
      rs += __shfl_xor(rs, 2);
      rs += __shfl_xor(rs, 4);
      rs += __shfl_xor(rs, 8);
      l_r[r] = l_r[r] * alpha[r] + rs;
    }
#pragma unroll
    for (int dt = 0; dt < 4; ++dt)
#pragma unroll
      for (int r = 0; r < 8; ++r) { o0[dt][r] *= alpha[r]; o1[dt][r] *= alpha[r]; }

    // P: C-layout -> bf16 -> per-wave LDS -> A-layout
#pragma unroll
    for (int r = 0; r < 8; ++r) {
      int row = r + 8 * hl;
      lds[row * 32 + mr]      = f2bf(s0[r]);
      lds[row * 32 + 16 + mr] = f2bf(s1[r]);
    }
    asm volatile("s_wait_dscnt 0x0" ::: "memory");
    ABu pa;
#pragma unroll
    for (int r = 0; r < 8; ++r) {
      int kk = a_koff(r, hl);
      pa.u[r] = *(const u32*)(lds + mr * 32 + kk);
    }

#pragma unroll
    for (int dt = 0; dt < 4; ++dt) {
      int d = dt * 16 + mr;
      ABu vb0, vb1;
#pragma unroll
      for (int r = 0; r < 8; ++r) {
        int key = hl * 16 + 2 * r;
        vb0.u[r] = (u32)ldsV0[key * 64 + d] | ((u32)ldsV0[(key + 1) * 64 + d] << 16);
        vb1.u[r] = (u32)ldsV1[key * 64 + d] | ((u32)ldsV1[(key + 1) * 64 + d] << 16);
      }
      o0[dt] = wmma_bf(pa, vb0, o0[dt]);
      o1[dt] = wmma_bf(pa, vb1, o1[dt]);
    }
    __syncthreads();
  }

  float inv[8];
#pragma unroll
  for (int r = 0; r < 8; ++r) inv[r] = 1.0f / l_r[r];
#pragma unroll
  for (int dt = 0; dt < 4; ++dt)
#pragma unroll
    for (int r = 0; r < 8; ++r) {
      int t = qbase + r + 8 * hl;
      size_t o = (bhT + t) * D_ + dt * 16 + mr;
      O0[o] = f2bf(o0[dt][r] * inv[r]);
      if (O1) O1[o] = f2bf(o1[dt][r] * inv[r]);
    }
}

// ---------------------------------------------------------------------------
// min/max over the 8 interval combos
// ---------------------------------------------------------------------------
struct RedArgs { const u16* yc[8]; u16 *ylo, *yhi; };

__global__ void reduce_mm(RedArgs g) {
  int i = blockIdx.x * 256 + threadIdx.x;
  if (i >= BTC_) return;
  float lo = 3e38f, hi = -3e38f;
#pragma unroll
  for (int c = 0; c < 8; ++c) {
    float v = bf2f(g.yc[c][i]);
    lo = fminf(lo, v);
    hi = fmaxf(hi, v);
  }
  g.ylo[i] = f2bf(lo);
  g.yhi[i] = f2bf(hi);
}

// ---------------------------------------------------------------------------
// Output projection: out[z][b,t,c'] = sum_c y_z[b,h(c),t,d(c)] * Wproj[c',c]
// Software-pipelined like qkv_gemm.
// grid=(C/64=12, BT/16=256, 3); block=128
// ---------------------------------------------------------------------------
struct ProjArgs { const u16* y[3]; const u16* wp; float* out; };

__device__ __forceinline__ void proj_load(const u16* Y, const u16* wp, size_t brow,
                                          int ab, int at, int kc, int hl,
                                          ABu& av, ABu& bw) {
#pragma unroll
  for (int r = 0; r < 8; ++r) {
    int c = kc * 32 + a_koff(r, hl);
    av.u[r] = *(const u32*)(Y + (((size_t)ab * H_ + (c >> 6)) * T_ + at) * D_ + (c & 63));
    int kb = kc * 32 + hl * 16 + 2 * r;
    bw.u[r] = *(const u32*)(wp + brow + kb);
  }
}

__global__ __launch_bounds__(128) void proj_gemm(ProjArgs g) {
  int lane = threadIdx.x & 31, wave = threadIdx.x >> 5;
  int hl = lane >> 4, mr = lane & 15;
  int cbase = blockIdx.x * 64 + wave * 16;
  int rbase = blockIdx.y * 16;
  int z = blockIdx.z;
  const u16* Y = g.y[z];

  v8f acc = {0,0,0,0,0,0,0,0};
  int row = rbase + mr;
  int ab = row >> 11, at = row & (T_ - 1);
  const size_t brow = (size_t)(cbase + mr) * C_;

  ABu av0, bw0, av1, bw1;
  proj_load(Y, g.wp, brow, ab, at, 0, hl, av0, bw0);

  const int NK = C_ / 32;        // 24 (even)
  for (int kc = 0; kc < NK; kc += 2) {
    proj_load(Y, g.wp, brow, ab, at, kc + 1, hl, av1, bw1);
    acc = wmma_bf(av0, bw0, acc);
    if (kc + 2 < NK)
      proj_load(Y, g.wp, brow, ab, at, kc + 2, hl, av0, bw0);
    acc = wmma_bf(av1, bw1, acc);
  }

  float* outp = g.out + (size_t)z * BTC_;
  int cc = cbase + mr;
#pragma unroll
  for (int r = 0; r < 8; ++r) {
    int rrow = rbase + r + 8 * hl;
    int b = rrow >> 11, t = rrow & (T_ - 1);
    outp[((size_t)b * T_ + t) * C_ + cc] = acc[r];
  }
}

// ---------------------------------------------------------------------------
extern "C" void kernel_launch(void* const* d_in, const int* in_sizes, int n_in,
                              void* d_out, int out_size, void* d_ws, size_t ws_size,
                              hipStream_t stream) {
  const float* x  = (const float*)d_in[0];
  const float* xl = (const float*)d_in[1];
  const float* xu = (const float*)d_in[2];
  const float* W  = (const float*)d_in[3];
  const float* Wp = (const float*)d_in[4];
  float* out = (float*)d_out;

  char* ws = (char*)d_ws;
  size_t off = 0;
  auto alloc = [&](size_t bytes) -> char* {
    char* p = ws + off;
    off += (bytes + 255) & ~(size_t)255;
    return p;
  };

  u16* xbf  = (u16*)alloc(BTC_ * 2);
  u16* xcbf = (u16*)alloc(BTC_ * 2);
  u16* xrbf = (u16*)alloc(BTC_ * 2);
  u16* wbf  = (u16*)alloc(W3_ * 2);
  u16* wabf = (u16*)alloc(W3_ * 2);
  u16* wpbf = (u16*)alloc(WP_ * 2);

  u16* qkv_nom[3]; u16* qkv_lo[3]; u16* qkv_hi[3];
  for (int s = 0; s < 3; ++s) qkv_nom[s] = (u16*)alloc(BTC_ * 2);
  for (int s = 0; s < 3; ++s) qkv_lo[s]  = (u16*)alloc(BTC_ * 2);
  for (int s = 0; s < 3; ++s) qkv_hi[s]  = (u16*)alloc(BTC_ * 2);

  u16* ynom = (u16*)alloc(BTC_ * 2);
  u16* yc[8];
  for (int c = 0; c < 8; ++c) yc[c] = (u16*)alloc(BTC_ * 2);
  u16* ylo = (u16*)alloc(BTC_ * 2);
  u16* yhi = (u16*)alloc(BTC_ * 2);

  prep_x<<<(BTC_ + 255) / 256, 256, 0, stream>>>(x, xl, xu, xbf, xcbf, xrbf);
  prep_w<<<(W3_ + 255) / 256, 256, 0, stream>>>(W, Wp, wbf, wabf, wpbf);

  QkvArgs qa;
  qa.xb = xbf; qa.xcb = xcbf; qa.xrb = xrbf; qa.wb = wbf; qa.wab = wabf;
  for (int s = 0; s < 3; ++s) { qa.nom[s] = qkv_nom[s]; qa.lo[s] = qkv_lo[s]; qa.hi[s] = qkv_hi[s]; }
  qkv_gemm<<<dim3(C3_ / 64, BT_ / 16), 128, 0, stream>>>(qa);

  AttnArgs aa;
  aa.qn = qkv_nom[0]; aa.kn = qkv_nom[1]; aa.vn = qkv_nom[2];
  aa.ql = qkv_lo[0];  aa.qh = qkv_hi[0];
  aa.kl = qkv_lo[1];  aa.kh = qkv_hi[1];
  aa.vl = qkv_lo[2];  aa.vh = qkv_hi[2];
  aa.ynom = ynom;
  for (int c = 0; c < 8; ++c) aa.yc[c] = yc[c];
  attn_fa<<<dim3(T_ / 64, B_ * H_, 5), 128, 0, stream>>>(aa);

  RedArgs ra;
  for (int c = 0; c < 8; ++c) ra.yc[c] = yc[c];
  ra.ylo = ylo; ra.yhi = yhi;
  reduce_mm<<<(BTC_ + 255) / 256, 256, 0, stream>>>(ra);

  ProjArgs pa;
  pa.y[0] = ynom; pa.y[1] = ylo; pa.y[2] = yhi;
  pa.wp = wpbf; pa.out = out;
  proj_gemm<<<dim3(C_ / 64, BT_ / 16, 3), 128, 0, stream>>>(pa);
}
